// AttentionAugmentedConv2d_8229157339494
// MI455X (gfx1250) — matrix operand order repaired and checked by reading, not compiled, not run on hardware
//
#include <hip/hip_runtime.h>
#include <hip/hip_bf16.h>

// Attention-Augmented Conv2d for MI455X (gfx1250), wave32 + WMMA f16->f32.
//   1) conv3x3 (implicit GEMM, WMMA, branchless im2col) -> out[:, 0:256]
//   2) conv3x3 (implicit GEMM, WMMA)                    -> ws.qkv (B,768,1024)
//   3) rel tables RX/RY[bh,i,m] = scale * q(i) . kr[m]  (VALU, tiny)
//   4) logits = QK^T + rel gathers; 32x64 block per wave (8 WMMA) -> ws.logits
//   5) row softmax in place (128MB, L2-resident: MI455X has 192MB L2)
//   6) O = P @ V^T; 16x32 block per wave (2 WMMA / k-step, P read once)
//   7) 1x1 conv (WMMA, K=256)                           -> out[:, 256:512]
// Workspace requirement ~168 MB.

typedef __attribute__((ext_vector_type(16))) _Float16 v16h;
typedef __attribute__((ext_vector_type(8)))  float    v8f;

#define CIN 128

// ---- WMMA fragment layouts (CDNA5 ISA §7.12.2) ----
// A (16x32 f16): lane L: M = L%16; g=L>=16; half t<8 -> K = 8g+t ; t>=8 -> K = 16+8g+(t-8)
//   => two contiguous 8-float runs at [8g, 8g+8) and [16+8g, 16+8g+8)
// B (32x16 f16): lane L: N = L%16; table order as A; code uses K = 16g + t for A and B (one 16-float run)
// C/D (16x16 f32, 8 VGPRs): elem r: M = r + 8g; N = L%16

static __device__ __forceinline__ v16h frag_two_runs(const float* run1, const float* run2) {
  float4 u0 = *(const float4*)(run1);
  float4 u1 = *(const float4*)(run1 + 4);
  float4 u2 = *(const float4*)(run2);
  float4 u3 = *(const float4*)(run2 + 4);
  v16h a;
  a[0]=(_Float16)u0.x;  a[1]=(_Float16)u0.y;  a[2]=(_Float16)u0.z;  a[3]=(_Float16)u0.w;
  a[4]=(_Float16)u1.x;  a[5]=(_Float16)u1.y;  a[6]=(_Float16)u1.z;  a[7]=(_Float16)u1.w;
  a[8]=(_Float16)u2.x;  a[9]=(_Float16)u2.y;  a[10]=(_Float16)u2.z; a[11]=(_Float16)u2.w;
  a[12]=(_Float16)u3.x; a[13]=(_Float16)u3.y; a[14]=(_Float16)u3.z; a[15]=(_Float16)u3.w;
  return a;
}

static __device__ __forceinline__ v16h frag_run16(const float* p) {
  return frag_two_runs(p, p + 8);
}

// ---------------- 1) / 2) conv3x3 as implicit GEMM ----------------
// C[M=Cout][N=B*1024] = W[M][K=Cin*9] * im2col[K][N], K order = (ci, r, s)
// so weight rows are contiguous -> float4 A loads. im2col is branchless:
// clamped-address load + select (no exec-mask churn, loads form clauses).
__global__ void conv3x3_wmma_kernel(const float* __restrict__ x,
                                    const float* __restrict__ w,
                                    const float* __restrict__ bias,
                                    float* __restrict__ out,
                                    int Cout, int outBatchCh, int outChanOff)
{
  const int wave = threadIdx.x >> 5;
  const int lane = threadIdx.x & 31;
  const int g    = lane >> 4;
  const int ln   = lane & 15;
  const int nTilesN = 256;                       // 4096/16
  const int tile = blockIdx.x * 8 + wave;
  if (tile >= (Cout >> 4) * nTilesN) return;
  const int mt = tile / nTilesN;
  const int nt = tile - mt * nTilesN;

  const int m  = mt * 16 + ln;
  const int n  = nt * 16 + ln;
  const int b  = n >> 10;
  const int p  = n & 1023;
  const int py = p >> 5;
  const int px = p & 31;
  const float* wrow = w + (size_t)m * (CIN * 9);
  const float* xb   = x + (size_t)b * (CIN * 1024);

  v8f acc = {0.f,0.f,0.f,0.f,0.f,0.f,0.f,0.f};
  for (int kb = 0; kb < CIN * 9; kb += 32) {
    v16h a = frag_run16(wrow + kb + g * 16);
    v16h bf;
    int k   = kb + g * 16;
    int ci  = k / 9;
    int rem = k - ci * 9;
    int r   = rem / 3;
    int s   = rem - r * 3;
    #pragma unroll
    for (int t = 0; t < 16; ++t) {
      int iy = py + r - 1;
      int ix = px + s - 1;
      bool ok = ((unsigned)iy < 32u) && ((unsigned)ix < 32u);
      int iyc = min(max(iy, 0), 31);
      int ixc = min(max(ix, 0), 31);
      float v = xb[(ci * 32 + iyc) * 32 + ixc];   // always in-bounds
      bf[t] = (_Float16)(ok ? v : 0.f);
      if (++s == 3) { s = 0; if (++r == 3) { r = 0; ++ci; } }
    }
    acc = __builtin_amdgcn_wmma_f32_16x16x32_f16(false, a, false, bf, (short)0, acc, false, false);
  }
  #pragma unroll
  for (int r = 0; r < 8; ++r) {
    int c = mt * 16 + r + g * 8;
    out[((size_t)b * outBatchCh + outChanOff + c) * 1024 + p] = acc[r] + bias[c];
  }
}

// ---------------- 3) relative-logit tables ----------------
// RX[bh,i,m] = scale * sum_d q[b,h,d,i] * kr_x[m,d]   (m in [0,63), dkh=32)
__global__ void rel_logits_kernel(const float* __restrict__ qkv,
                                  const float* __restrict__ krx,
                                  const float* __restrict__ kry,
                                  float* __restrict__ RX,
                                  float* __restrict__ RY)
{
  __shared__ float qs[32];
  const int bh = blockIdx.x >> 10;
  const int i  = blockIdx.x & 1023;
  const int b  = bh >> 3;
  const int h  = bh & 7;
  const float scale = 0.17677669529663687f;      // 32^-0.5
  if (threadIdx.x < 32)
    qs[threadIdx.x] = qkv[((size_t)b * 768 + h * 32 + threadIdx.x) * 1024 + i] * scale;
  __syncthreads();
  const int tid = threadIdx.x;
  if (tid < 63) {
    const float* kr = krx + tid * 32;
    float s = 0.f;
    #pragma unroll
    for (int d = 0; d < 32; ++d) s += qs[d] * kr[d];
    RX[((size_t)bh * 1024 + i) * 63 + tid] = s;
  } else if (tid >= 64 && tid < 127) {
    const int m = tid - 64;
    const float* kr = kry + m * 32;
    float s = 0.f;
    #pragma unroll
    for (int d = 0; d < 32; ++d) s += qs[d] * kr[d];
    RY[((size_t)bh * 1024 + i) * 63 + m] = s;
  }
}

// ---------------- 4) logits = QK^T + rel_x + rel_y ----------------
// Per (b,h): 1024x1024, K=32 (single WMMA depth). Each wave computes a 32x64
// output block: 2 A-frags x 4 B-frags -> 8 WMMAs, ~12 frag loads per WMMA.
// rel_x(i,j) = RX[i, (j%32)-(i%32)+31]
// rel_y(i,j) = RY[iT, (j/32)-(i%32)+31],  iT = (i%32)*32 + i/32  (ref's transpose bug)
__global__ void logits_wmma_kernel(const float* __restrict__ qkv,
                                   const float* __restrict__ RX,
                                   const float* __restrict__ RY,
                                   float* __restrict__ logits)
{
  const int wave = threadIdx.x >> 5;
  const int lane = threadIdx.x & 31;
  const int g    = lane >> 4;
  const int ln   = lane & 15;
  const int tile = blockIdx.x * 8 + wave;        // 32 bh * 512 blocks
  const int bh = tile >> 9;
  const int t2 = tile & 511;
  const int it2 = t2 >> 4;                       // 0..31 : i block of 32
  const int jb  = t2 & 15;                       // 0..15 : j block of 64
  const int b  = bh >> 3;
  const int h  = bh & 7;
  const float scale = 0.17677669529663687f;

  const int i0 = it2 * 32;
  const int j0 = jb * 64;
  const float* qb = qkv + ((size_t)b * 768 + h * 32) * 1024;
  const float* kk = qkv + ((size_t)b * 768 + 256 + h * 32) * 1024;

  v16h aA[2];
  #pragma unroll
  for (int u = 0; u < 2; ++u) {
    const int i = i0 + u * 16 + ln;
    #pragma unroll
    for (int t = 0; t < 16; ++t) {
      int dA = g * 16 + t;
      aA[u][t] = (_Float16)(qb[dA * 1024 + i] * scale);
    }
  }
  v16h bB[4];
  #pragma unroll
  for (int vv = 0; vv < 4; ++vv) {
    const int j = j0 + vv * 16 + ln;
    #pragma unroll
    for (int t = 0; t < 16; ++t) {
      int dB = g * 16 + t;
      bB[vv][t] = (_Float16)(kk[dB * 1024 + j]);
    }
  }

  #pragma unroll
  for (int u = 0; u < 2; ++u) {
    #pragma unroll
    for (int vv = 0; vv < 4; ++vv) {
      v8f acc = {0.f,0.f,0.f,0.f,0.f,0.f,0.f,0.f};
      acc = __builtin_amdgcn_wmma_f32_16x16x32_f16(false, aA[u], false, bB[vv],
                                                   (short)0, acc, false, false);
      #pragma unroll
      for (int r = 0; r < 8; ++r) {
        int ie = i0 + u * 16 + r + g * 8;
        int je = j0 + vv * 16 + ln;
        int ix = ie & 31, iy = ie >> 5;
        int mx = (je & 31) - ix + 31;
        int my = (je >> 5) - ix + 31;
        int iT = ix * 32 + iy;
        float rx = RX[((size_t)bh * 1024 + ie) * 63 + mx];
        float ry = RY[((size_t)bh * 1024 + iT) * 63 + my];
        logits[(size_t)bh * 1048576 + (size_t)ie * 1024 + je] = acc[r] + rx + ry;
      }
    }
  }
}

// ---------------- 5) row softmax (1024 wide, in place) ----------------
__global__ void softmax_kernel(float* __restrict__ logits)
{
  __shared__ float red[256];
  float* ptr = logits + (size_t)blockIdx.x * 1024;
  const int tid = threadIdx.x;
  float v[4];
  float mx = -3.402823466e38f;
  #pragma unroll
  for (int k = 0; k < 4; ++k) { v[k] = ptr[tid + k * 256]; mx = fmaxf(mx, v[k]); }
  red[tid] = mx; __syncthreads();
  for (int s = 128; s > 0; s >>= 1) { if (tid < s) red[tid] = fmaxf(red[tid], red[tid + s]); __syncthreads(); }
  mx = red[0]; __syncthreads();
  float sum = 0.f;
  #pragma unroll
  for (int k = 0; k < 4; ++k) { v[k] = __expf(v[k] - mx); sum += v[k]; }
  red[tid] = sum; __syncthreads();
  for (int s = 128; s > 0; s >>= 1) { if (tid < s) red[tid] += red[tid + s]; __syncthreads(); }
  const float inv = 1.0f / red[0];
  #pragma unroll
  for (int k = 0; k < 4; ++k) ptr[tid + k * 256] = v[k] * inv;
}

// ---------------- 6) O = P @ V^T  (per bh: 1024 x 32, K = 1024) ----------------
// One wave per 16-row block covers BOTH 16-wide d tiles: each A-frag (P) feeds
// two WMMAs, so the 128MB P matrix is streamed exactly once.
// O stored row-major (i,d): by the reference's raw-reshape bug this IS the
// channel-major (256,1024) layout the 1x1 conv consumes.
__global__ void av_wmma_kernel(const float* __restrict__ qkv,
                               const float* __restrict__ P,
                               float* __restrict__ O)
{
  const int wave = threadIdx.x >> 5;
  const int lane = threadIdx.x & 31;
  const int g    = lane >> 4;
  const int ln   = lane & 15;
  const int tile = blockIdx.x * 8 + wave;        // 32 bh * 64 i-tiles
  const int bh = tile >> 6;
  const int it = tile & 63;
  const int b  = bh >> 3;
  const int h  = bh & 7;

  const int i = it * 16 + ln;
  const float* prow   = P + ((size_t)bh * 1024 + i) * 1024;
  const float* vbase0 = qkv + ((size_t)b * 768 + 512 + h * 32 + ln) * 1024;        // d = ln
  const float* vbase1 = qkv + ((size_t)b * 768 + 512 + h * 32 + 16 + ln) * 1024;   // d = 16+ln

  v8f acc0 = {0.f,0.f,0.f,0.f,0.f,0.f,0.f,0.f};
  v8f acc1 = {0.f,0.f,0.f,0.f,0.f,0.f,0.f,0.f};
  for (int kb = 0; kb < 1024; kb += 32) {
    v16h a  = frag_run16(prow + kb + g * 16);
    v16h b0 = frag_run16(vbase0 + kb + g * 16);
    v16h b1 = frag_run16(vbase1 + kb + g * 16);
    acc0 = __builtin_amdgcn_wmma_f32_16x16x32_f16(false, a, false, b0, (short)0, acc0, false, false);
    acc1 = __builtin_amdgcn_wmma_f32_16x16x32_f16(false, a, false, b1, (short)0, acc1, false, false);
  }
  #pragma unroll
  for (int r = 0; r < 8; ++r) {
    int ie = it * 16 + r + g * 8;
    O[(size_t)bh * 32768 + ie * 32 + ln]      = acc0[r];
    O[(size_t)bh * 32768 + ie * 32 + 16 + ln] = acc1[r];
  }
}

// ---------------- 7) 1x1 conv: out[:,256:512] = att_w @ O + att_b ----------------
__global__ void conv1x1_wmma_kernel(const float* __restrict__ O,
                                    const float* __restrict__ w,   // (256,256)
                                    const float* __restrict__ bias,
                                    float* __restrict__ out)
{
  const int wave = threadIdx.x >> 5;
  const int lane = threadIdx.x & 31;
  const int g    = lane >> 4;
  const int ln   = lane & 15;
  const int tile = blockIdx.x * 8 + wave;        // 4 * 16 * 64 tiles
  const int b  = tile >> 10;
  const int t2 = tile & 1023;
  const int mt = t2 >> 6;
  const int pt = t2 & 63;

  const int co = mt * 16 + ln;
  const int p  = pt * 16 + ln;
  const float* wrow = w + (size_t)co * 256;
  const float* Ob   = O + (size_t)b * 262144;

  v8f acc = {0.f,0.f,0.f,0.f,0.f,0.f,0.f,0.f};
  for (int kb = 0; kb < 256; kb += 32) {
    v16h a = frag_run16(wrow + kb + g * 16);
    v16h bf;
    #pragma unroll
    for (int t = 0; t < 16; ++t) {
      int ci = kb + g * 16 + t;
      bf[t] = (_Float16)Ob[(size_t)ci * 1024 + p];
    }
    acc = __builtin_amdgcn_wmma_f32_16x16x32_f16(false, a, false, bf, (short)0, acc, false, false);
  }
  #pragma unroll
  for (int r = 0; r < 8; ++r) {
    int coe = mt * 16 + r + g * 8;
    out[((size_t)b * 512 + 256 + coe) * 1024 + p] = acc[r] + bias[coe];
  }
}

extern "C" void kernel_launch(void* const* d_in, const int* in_sizes, int n_in,
                              void* d_out, int out_size, void* d_ws, size_t ws_size,
                              hipStream_t stream) {
  const float* x      = (const float*)d_in[0];
  const float* conv_w = (const float*)d_in[1];
  const float* conv_b = (const float*)d_in[2];
  const float* qkv_w  = (const float*)d_in[3];
  const float* qkv_b  = (const float*)d_in[4];
  const float* att_w  = (const float*)d_in[5];
  const float* att_b  = (const float*)d_in[6];
  const float* kr_x   = (const float*)d_in[7];
  const float* kr_y   = (const float*)d_in[8];
  float* out = (float*)d_out;
  float* ws  = (float*)d_ws;

  float* qkv    = ws;                                   // 4*768*1024     = 3,145,728 f
  float* RX     = qkv + 3145728;                        // 32*1024*63    = 2,064,384 f
  float* RY     = RX  + 2064384;                        // 2,064,384 f
  float* O      = RY  + 2064384;                        // 4*256*1024    = 1,048,576 f
  float* logits = O   + 1048576;                        // 32*1024*1024  = 33,554,432 f

  conv3x3_wmma_kernel<<<512,  256, 0, stream>>>(x, conv_w, conv_b, out, 256, 512, 0);
  conv3x3_wmma_kernel<<<1536, 256, 0, stream>>>(x, qkv_w,  qkv_b,  qkv, 768, 768, 0);
  rel_logits_kernel  <<<32768, 128, 0, stream>>>(qkv, kr_x, kr_y, RX, RY);
  logits_wmma_kernel <<<2048,  256, 0, stream>>>(qkv, RX, RY, logits);
  softmax_kernel     <<<32768, 256, 0, stream>>>(logits);
  av_wmma_kernel     <<<256,   256, 0, stream>>>(qkv, logits, O);
  conv1x1_wmma_kernel<<<512,   256, 0, stream>>>(O, att_w, att_b, out);
}
